// GTRNN_2413771620939
// MI455X (gfx1250) — matrix operand order repaired and checked by reading, not compiled, not run on hardware
//
#include <hip/hip_runtime.h>
#include <hip/hip_bf16.h>

// ---------------------------------------------------------------------------
// GT-RNN forward as 8 fused GEMMs on CDNA5 WMMA (bf16 in, f32 accumulate).
// ---------------------------------------------------------------------------

typedef __bf16 bf16_t;
typedef bf16_t v16bf __attribute__((ext_vector_type(16)));
typedef bf16_t v8bf  __attribute__((ext_vector_type(8)));
typedef float  v8f   __attribute__((ext_vector_type(8)));

#define HID    512
#define SEQ    4096
#define BATCH  8
#define NTOK   (SEQ * BATCH)     // 32768 tokens

#define WG_THREADS 256           // 8 waves (wave32)
#define M_TILE     128           // rows per workgroup (16 per wave)
#define N_TILE     64            // cols per workgroup (4 WMMA subtiles per wave)
#define K_STAGE    256           // K staged in LDS at a time (2 stages per pass)
#define K_CHUNK    32            // K per v_wmma_f32_16x16x32_bf16
#define LDS_STRIDE 264           // bf16/row: 528 B = 16B-aligned; start bank 4*l -> conflict-free

static __device__ __forceinline__ bf16_t f2b(float f) { return (bf16_t)f; }

// branch-free tanh: 1 - 2/(exp2(2*log2e*x)+1); saturates to +/-1, no EXEC games
static __device__ __forceinline__ float tanh_fast(float x) {
    float e = __builtin_amdgcn_exp2f(x * 2.88539008177792681472f); // e^(2x)
    return 1.0f - 2.0f * __builtin_amdgcn_rcpf(e + 1.0f);
}

// out[M x 512] = act( A1 @ W1^T (+ A2 @ W2^T) + bias1 (+ bias2) )
// W matrices are [512 out x 512 in] row-major (torch Linear layout),
// so B = W^T and column n of B is row n of W: contiguous loads.
__global__ __launch_bounds__(WG_THREADS)
void gtrnn_gemm_bf16wmma(const float* __restrict__ A1, const float* __restrict__ W1,
                         const float* __restrict__ A2, const float* __restrict__ W2,
                         const float* __restrict__ bias1, const float* __restrict__ bias2,
                         float* __restrict__ out, int do_tanh, int transpose_out)
{
    __shared__ __align__(16) bf16_t Bs[N_TILE][LDS_STRIDE];   // ~33 KB

    const int tid   = threadIdx.x;
    const int wave  = tid >> 5;
    const int lane  = tid & 31;
    const int lmod  = lane & 15;       // row-in-subtile / col-in-subtile
    const int lhalf = lane >> 4;       // 0 or 1

    const int m_base = blockIdx.x * M_TILE;
    const int n_base = blockIdx.y * N_TILE;

    v8f acc[4];
#pragma unroll
    for (int t = 0; t < 4; ++t)
#pragma unroll
        for (int r = 0; r < 8; ++r) acc[t][r] = 0.0f;

    // A-fragment lane addressing (16x32 bf16 A-matrix layout, ISA 7.12.2):
    //  lanes 0-15 : K = {0..7, 16..23};  lanes 16-31 : K = {8..15, 24..31}
    const int    klo   = lhalf * 8;
    const size_t m_row = (size_t)(m_base + wave * 16 + lmod);

    // cooperative B staging: 64 rows x K_STAGE, 8 f32 per thread per step
    const int wrow = tid >> 2;          // 0..63
    const int wcol = (tid & 3) << 3;    // 0,8,16,24

    const int npass = (A2 != nullptr) ? 2 : 1;
    for (int pass = 0; pass < npass; ++pass) {
        const float* __restrict__ A = pass ? A2 : A1;
        const float* __restrict__ W = pass ? W2 : W1;

        for (int ks = 0; ks < HID; ks += K_STAGE) {
            // ---- stage 64 x K_STAGE weight slab (f32 -> bf16) into LDS ----
#pragma unroll
            for (int j = 0; j < K_STAGE; j += 32) {
                const float* wsrc = W + (size_t)(n_base + wrow) * HID + ks + wcol + j;
                float4 w0 = ((const float4*)wsrc)[0];
                float4 w1 = ((const float4*)wsrc)[1];
                v8bf pk;
                pk[0] = f2b(w0.x); pk[1] = f2b(w0.y); pk[2] = f2b(w0.z); pk[3] = f2b(w0.w);
                pk[4] = f2b(w1.x); pk[5] = f2b(w1.y); pk[6] = f2b(w1.z); pk[7] = f2b(w1.w);
                *(v8bf*)&Bs[wrow][wcol + j] = pk;
            }
            __syncthreads();

            // ---- barrier-free K loop over the staged slab ----
#pragma unroll 2
            for (int kc = 0; kc < K_STAGE; kc += K_CHUNK) {
                // per-wave A fragment from global (float4 loads, cvt to bf16)
                const float* arow = A + m_row * HID + ks + kc + klo;
                float4 a0 = ((const float4*)arow)[0];
                float4 a1 = ((const float4*)arow)[1];
                float4 a2 = ((const float4*)(arow + 16))[0];
                float4 a3 = ((const float4*)(arow + 16))[1];
                v16bf af;
                af[0]  = f2b(a0.x); af[1]  = f2b(a0.y); af[2]  = f2b(a0.z); af[3]  = f2b(a0.w);
                af[4]  = f2b(a1.x); af[5]  = f2b(a1.y); af[6]  = f2b(a1.z); af[7]  = f2b(a1.w);
                af[8]  = f2b(a2.x); af[9]  = f2b(a2.y); af[10] = f2b(a2.z); af[11] = f2b(a2.w);
                af[12] = f2b(a3.x); af[13] = f2b(a3.y); af[14] = f2b(a3.z); af[15] = f2b(a3.w);

                // preload ALL 4 B fragments (8 ds_load_b128, one wait), then
                // issue the 4 WMMAs back-to-back on the XDL pipe
                v16bf bf[4];
#pragma unroll
                for (int t = 0; t < 4; ++t) {
                    const v8bf* bp = (const v8bf*)&Bs[t * 16 + lmod][kc + lhalf * 8];
                    v8bf blo = bp[0];
                    v8bf bhi = bp[2];
                    bf[t] = __builtin_shufflevector(blo, bhi,
                                0,1,2,3,4,5,6,7,8,9,10,11,12,13,14,15);
                }
#pragma unroll
                for (int t = 0; t < 4; ++t) {
                    acc[t] = __builtin_amdgcn_wmma_f32_16x16x32_bf16(
                                /*neg_a=*/false, af, /*neg_b=*/false, bf[t],
                                /*c_mod=*/(short)0, acc[t],
                                /*reuse_a=*/false, /*reuse_b=*/false);
                }
            }
            __syncthreads();
        }
    }

    // ---- epilogue: bias, tanh, store (C/D layout: VGPR r -> M=r+8*lhalf, N=lmod)
    const int row0 = m_base + wave * 16 + lhalf * 8;
#pragma unroll
    for (int t = 0; t < 4; ++t) {
        const int col = n_base + t * 16 + lmod;
        float bsum = bias1[col] + (bias2 ? bias2[col] : 0.0f);
#pragma unroll
        for (int r = 0; r < 8; ++r) {
            float v = acc[t][r] + bsum;
            if (do_tanh) v = tanh_fast(v);
            const int m = row0 + r;               // token index g = b*SEQ + s
            size_t oidx;
            if (transpose_out) {
                const int bi = m >> 12;           // / SEQ
                const int si = m & (SEQ - 1);     // % SEQ
                oidx = (size_t)(si * BATCH + bi) * HID + col;
            } else {
                oidx = (size_t)m * HID + col;
            }
            out[oidx] = v;
        }
    }
}

// hs_final = h2 at the last timestep of every batch row: 8 x 512 gather
__global__ void gtrnn_hsfinal(const float* __restrict__ h2, float* __restrict__ out)
{
    int i = blockIdx.x * blockDim.x + threadIdx.x;     // 0..4095
    if (i >= BATCH * HID) return;
    int b = i >> 9;
    int c = i & (HID - 1);
    out[i] = h2[(size_t)(b * SEQ + (SEQ - 1)) * HID + c];
}

extern "C" void kernel_launch(void* const* d_in, const int* in_sizes, int n_in,
                              void* d_out, int out_size, void* d_ws, size_t ws_size,
                              hipStream_t stream)
{
    (void)in_sizes; (void)n_in; (void)out_size; (void)ws_size;

    const float* xs     = (const float*)d_in[0];
    const float* W_xh0  = (const float*)d_in[1];
    const float* b_xh0  = (const float*)d_in[2];
    // W_hh0 (d_in[3]) multiplies hs==0, only its bias contributes
    const float* b_hh0  = (const float*)d_in[4];
    const float* W_hy0  = (const float*)d_in[5];
    const float* b_hy0  = (const float*)d_in[6];
    const float* W_xh_h = (const float*)d_in[7];   // [2,512,512]
    const float* b_xh_h = (const float*)d_in[8];   // [2,512]
    const float* W_hh_h = (const float*)d_in[9];
    const float* b_hh_h = (const float*)d_in[10];
    const float* W_hy_h = (const float*)d_in[11];
    const float* b_hy_h = (const float*)d_in[12];

    float* out  = (float*)d_out;
    float* buf0 = (float*)d_ws;                        // 64 MB each
    float* buf1 = buf0 + (size_t)NTOK * HID;
    float* buf2 = buf1 + (size_t)NTOK * HID;

    const size_t WSTRIDE = (size_t)HID * HID;          // per-layer weight stride
    dim3 grid(NTOK / M_TILE, HID / N_TILE);            // (256, 8)
    dim3 block(WG_THREADS);

    // stage0: h0 = tanh(X @ W_xh0^T + b_xh0 + b_hh0)          -> buf0
    gtrnn_gemm_bf16wmma<<<grid, block, 0, stream>>>(
        xs, W_xh0, nullptr, nullptr, b_xh0, b_hh0, buf0, 1, 0);

    // stage1: y0 = h0 @ W_hy0^T + b_hy0                       -> buf1
    gtrnn_gemm_bf16wmma<<<grid, block, 0, stream>>>(
        buf0, W_hy0, nullptr, nullptr, b_hy0, nullptr, buf1, 0, 0);

    // stage2: h1 = tanh(h0@W_hh_h0^T + y0@W_xh_h0^T + biases) -> buf2
    gtrnn_gemm_bf16wmma<<<grid, block, 0, stream>>>(
        buf0, W_hh_h, buf1, W_xh_h, b_hh_h, b_xh_h, buf2, 1, 0);

    // stage3: y1 = h1 @ W_hy_h0^T + b_hy_h0                   -> buf0 (h0 dead)
    gtrnn_gemm_bf16wmma<<<grid, block, 0, stream>>>(
        buf2, W_hy_h, nullptr, nullptr, b_hy_h, nullptr, buf0, 0, 0);

    // stage4: h2 = tanh(h1@W_hh_h1^T + y1@W_xh_h1^T + biases) -> buf1 (y0 dead)
    gtrnn_gemm_bf16wmma<<<grid, block, 0, stream>>>(
        buf2, W_hh_h + WSTRIDE, buf0, W_xh_h + WSTRIDE,
        b_hh_h + HID, b_xh_h + HID, buf1, 1, 0);

    // stage5: y2 = h2 @ W_hy_h1^T + b_hy_h1, [B,S]->[S,B] transpose -> d_out
    gtrnn_gemm_bf16wmma<<<grid, block, 0, stream>>>(
        buf1, W_hy_h + WSTRIDE, nullptr, nullptr,
        b_hy_h + HID, nullptr, out, 0, 1);

    // hs_final: last timestep of h2 per batch -> d_out tail
    gtrnn_hsfinal<<<(BATCH * HID + 255) / 256, 256, 0, stream>>>(
        buf1, out + (size_t)NTOK * HID);
}
